// ELMO_T_86801289052566
// MI455X (gfx1250) — matrix operand order repaired and checked by reading, not compiled, not run on hardware
//
#include <hip/hip_runtime.h>

// ---------------------------------------------------------------------------
// ELMo-style stacked LSTM for MI455X (gfx1250, wave32, WMMA).
// B=256, S=128, H=512, gates=4H=2048.
// - bf16 V_WMMA_F32_16X16X32_BF16 for all x@Wih^T + h@Whh^T GEMMs
// - GLOBAL_LOAD_ASYNC_TO_LDS_B128 (ASYNCcnt) double-buffered staging: next
//   K-tile streams into LDS while current tile runs 16 WMMAs
// - fused sigmoid/tanh/c/h epilogue; one kernel per timestep (graph-replayed)
// - softmax over a size-1 axis == 1.0 exactly -> pooling is a plain seq sum
// ---------------------------------------------------------------------------

typedef __bf16 bf16_t;
typedef __bf16 v16bf __attribute__((ext_vector_type(16)));
typedef __bf16 v8bf  __attribute__((ext_vector_type(8)));
typedef float  v8f   __attribute__((ext_vector_type(8)));

#define HDIM   512
#define SEQ    128
#define BATCH  256
#define BM     128   // batch tile per workgroup
#define BJ     64    // hidden tile per workgroup (x4 gates)
#define BK     32    // K depth per WMMA
#define AP32   36    // LDS pitch for f32 A tile (144B: 16B-aligned, bank-clean)
#define WPITCH 40    // LDS pitch for bf16 W tile (80B: 16B-aligned, bank-clean)

struct Ptrs16 { const float* p[16]; };

struct StepArgs {
  const float*  x;       long long xbs;     // x_t base (already time-offset), batch stride
  const float*  h_in;                        // 256x512 dense
  const float*  c_in;                        // 256x512 dense
  const bf16_t* Wih;                         // 2048x512 bf16
  const bf16_t* Whh;                         // 2048x512 bf16
  const float*  bih;
  const float*  bhh;
  float*        h_out;                       // 256x512 dense
  float*        c_out;                       // nullable (decoder_back keeps c0)
  float*        seq_out; long long seq_bs;   // nullable, time-offset
  const float*  res;     long long res_bs;   // nullable residual add, time-offset
  const float*  ref;     long long ref_bs;   // nullable: accumulate (ref-h)^2
  float*        sq_accum;                    // nullable
};

// CDNA5 async global->LDS copy (no VGPR round-trip, tracked by ASYNCcnt).
__device__ __forceinline__ void async_copy_b128(unsigned lds_off, const void* gaddr) {
  asm volatile("global_load_async_to_lds_b128 %0, %1, off"
               :: "v"(lds_off), "v"(gaddr) : "memory");
}
__device__ __forceinline__ void wait_async0() {
  asm volatile("s_wait_asynccnt 0" ::: "memory");
}

__global__ void k_cvt_weights(Ptrs16 src, bf16_t* dst) {
  size_t i = (size_t)blockIdx.x * blockDim.x + threadIdx.x;
  const size_t total  = (size_t)16 << 20;                  // 16 matrices of 2^20
  const size_t stride = (size_t)gridDim.x * blockDim.x;
  for (; i < total; i += stride) {
    size_t mat = i >> 20;
    size_t off = i & (((size_t)1 << 20) - 1);
    dst[i] = (bf16_t)src.p[mat][off];
  }
}

__global__ void k_fill0(float* p, int n) {
  int i = blockIdx.x * blockDim.x + threadIdx.x;
  for (; i < n; i += gridDim.x * blockDim.x) p[i] = 0.f;
}

__device__ __forceinline__ float sigm(float x) { return 1.f / (1.f + __expf(-x)); }

// One LSTM timestep: g = x@Wih^T + h@Whh^T (+biases), then gate math.
// Grid (2, 8): M-tiles of 128, hidden(j)-tiles of 64. 256 threads = 8 wave32.
__global__ __launch_bounds__(256) void k_lstm_step(StepArgs a) {
  __shared__ float  Af [2][BM * AP32];          // 2 x 128x32 f32  (18KB each)
  __shared__ bf16_t Wsh[2][4 * BJ * WPITCH];    // 2 x 256x32 bf16 (20KB each)
  __shared__ float  red[8];

  const int tid  = threadIdx.x;
  const int lane = tid & 31;
  const int wave = tid >> 5;
  const int half = lane >> 4;     // 0: lanes 0-15, 1: lanes 16-31
  const int l15  = lane & 15;
  const int mt   = blockIdx.x;    // 0..1
  const int jt   = blockIdx.y;    // 0..7
  const int wm   = (wave & 3) * 32;   // wave M offset within tile
  const int wj   = (wave >> 2) * 32;  // wave J offset within tile

  v8f acc[4][2][2];
  const v8f vz = {0.f,0.f,0.f,0.f,0.f,0.f,0.f,0.f};
  for (int q = 0; q < 4; ++q)
    for (int mi = 0; mi < 2; ++mi)
      for (int ji = 0; ji < 2; ++ji) acc[q][mi][ji] = vz;

  // 32 K-tiles total: iter 0..15 = x-phase (Wih), 16..31 = h-phase (Whh)
  auto stage = [&](int iter, int buf) {
    const int phase = iter >> 4;
    const int kk    = (iter & 15) * BK;
    const float*    Ab   = phase ? a.h_in : a.x;
    const long long Astr = phase ? (long long)HDIM : a.xbs;
    const bf16_t*   Wb   = phase ? a.Whh : a.Wih;
    #pragma unroll
    for (int it = 0; it < 4; ++it) {              // A: 128 rows x 32 f32
      int idx = tid + 256 * it;                   // 0..1023
      int row = idx >> 3;                         // 0..127
      int col = (idx & 7) * 4;                    // 0..28
      const float* gp = Ab + (long long)(mt * BM + row) * Astr + (kk + col);
      async_copy_b128((unsigned)(size_t)&Af[buf][row * AP32 + col], gp);
    }
    #pragma unroll
    for (int it = 0; it < 4; ++it) {              // W: 4 gates x 64 rows x 32 bf16
      int idx = tid + 256 * it;                   // 0..1023
      int row = idx >> 2;                         // 0..255
      int c8  = (idx & 3) * 8;                    // 0,8,16,24
      int n   = (row >> 6) * 512 + jt * BJ + (row & 63);
      const bf16_t* gp = Wb + (size_t)n * HDIM + (kk + c8);
      async_copy_b128((unsigned)(size_t)&Wsh[buf][row * WPITCH + c8], gp);
    }
  };

  stage(0, 0);
  wait_async0();
  __syncthreads();

  for (int iter = 0; iter < 32; ++iter) {
    const int buf = iter & 1;
    if (iter < 31) stage(iter + 1, buf ^ 1);   // overlap next tile with compute

    // A fragments (16x32): lanes 0-15 hold K{0..7,16..23}, 16-31 K{8..15,24..31}
    // convert f32->bf16 at fragment build (v_cvt_pk co-executes with WMMA pipe)
    v16bf af[2];
    #pragma unroll
    for (int mi = 0; mi < 2; ++mi) {
      const float* p = &Af[buf][(wm + mi * 16 + l15) * AP32 + half * 8];
      float4 f0 = *(const float4*)(p);
      float4 f1 = *(const float4*)(p + 4);
      float4 f2 = *(const float4*)(p + 16);
      float4 f3 = *(const float4*)(p + 20);
      v16bf v;
      v[0]  = (bf16_t)f0.x; v[1]  = (bf16_t)f0.y; v[2]  = (bf16_t)f0.z; v[3]  = (bf16_t)f0.w;
      v[4]  = (bf16_t)f1.x; v[5]  = (bf16_t)f1.y; v[6]  = (bf16_t)f1.z; v[7]  = (bf16_t)f1.w;
      v[8]  = (bf16_t)f2.x; v[9]  = (bf16_t)f2.y; v[10] = (bf16_t)f2.z; v[11] = (bf16_t)f2.w;
      v[12] = (bf16_t)f3.x; v[13] = (bf16_t)f3.y; v[14] = (bf16_t)f3.z; v[15] = (bf16_t)f3.w;
      af[mi] = v;
    }
    // B fragments (32x16): lanes 0-15 K{0..7,16..23}, lanes 16-31 K{8..15,24..31}, N=lane&15
    #pragma unroll
    for (int q = 0; q < 4; ++q) {
      #pragma unroll
      for (int ji = 0; ji < 2; ++ji) {
        const bf16_t* p = &Wsh[buf][(q * BJ + wj + ji * 16 + l15) * WPITCH + half * 8];
        v8bf b0 = *(const v8bf*)p;
        v8bf b1 = *(const v8bf*)(p + 16);
        v16bf bf = __builtin_shufflevector(b0, b1, 0,1,2,3,4,5,6,7,8,9,10,11,12,13,14,15);
        acc[q][0][ji] = __builtin_amdgcn_wmma_f32_16x16x32_bf16(
            false, af[0], false, bf, (short)0, acc[q][0][ji], false, false);
        acc[q][1][ji] = __builtin_amdgcn_wmma_f32_16x16x32_bf16(
            false, af[1], false, bf, (short)0, acc[q][1][ji], false, false);
      }
    }

    wait_async0();     // next buffer's async copies have landed in LDS
    __syncthreads();   // all waves done with this buffer before it is re-staged
  }

  // ---- fused LSTM gate epilogue ----
  // C/D layout: VGPR r -> M = r + half*8 (within 16x16 tile), N = lane&15
  float sq = 0.f;
  #pragma unroll
  for (int mi = 0; mi < 2; ++mi) {
    #pragma unroll
    for (int r = 0; r < 8; ++r) {
      int m = mt * BM + wm + mi * 16 + r + half * 8;
      #pragma unroll
      for (int ji = 0; ji < 2; ++ji) {
        int j = jt * BJ + wj + ji * 16 + l15;
        float gi = acc[0][mi][ji][r] + a.bih[j]        + a.bhh[j];
        float gf = acc[1][mi][ji][r] + a.bih[512 + j]  + a.bhh[512 + j];
        float gg = acc[2][mi][ji][r] + a.bih[1024 + j] + a.bhh[1024 + j];
        float go = acc[3][mi][ji][r] + a.bih[1536 + j] + a.bhh[1536 + j];
        float cn = sigm(gf) * a.c_in[(size_t)m * HDIM + j] + sigm(gi) * tanhf(gg);
        float hn = sigm(go) * tanhf(cn);
        if (a.c_out) a.c_out[(size_t)m * HDIM + j] = cn;
        a.h_out[(size_t)m * HDIM + j] = hn;
        if (a.seq_out) {
          float v = hn;
          if (a.res) v += a.res[(long long)m * a.res_bs + j];
          a.seq_out[(long long)m * a.seq_bs + j] = v;
        }
        if (a.ref) { float d = a.ref[(long long)m * a.ref_bs + j] - hn; sq += d * d; }
      }
    }
  }
  if (a.ref && a.sq_accum) {
    for (int off = 16; off > 0; off >>= 1) sq += __shfl_down(sq, off, 32);
    if (lane == 0) red[wave] = sq;
    __syncthreads();
    if (tid == 0) {
      float s = 0.f;
      for (int w = 0; w < 8; ++w) s += red[w];
      atomicAdd(a.sq_accum, s);
    }
  }
}

// pooled[b,:] = sum_s (x + e3 + b3_reversed)   (softmax over singleton axis == 1)
__global__ void k_pool(const float* x, const float* e3, const float* b3, float* pooled) {
  int b = blockIdx.x;
  const size_t base = (size_t)b * SEQ * HDIM;
  for (int j = threadIdx.x; j < HDIM; j += blockDim.x) {
    float s = 0.f;
    for (int t = 0; t < SEQ; ++t) {
      s += x [base + (size_t)t * HDIM + j]
         + e3[base + (size_t)t * HDIM + j]
         + b3[base + (size_t)(SEQ - 1 - t) * HDIM + j];
    }
    pooled[(size_t)b * HDIM + j] = s;
  }
}

__global__ void k_out(const float* pooled, const float* ow, const float* ob, float* out) {
  int b = blockIdx.x;
  int o = threadIdx.x;
  float s = ob[o];
  const float* pr = pooled + (size_t)b * HDIM;
  const float* wr = ow + (size_t)o * HDIM;
  for (int k = 0; k < HDIM; ++k) s += pr[k] * wr[k];
  out[(size_t)b * 256 + o] = s;
}

__global__ void k_finalize(const float* acc, float* dst) {
  if (threadIdx.x == 0 && blockIdx.x == 0)
    dst[0] = (acc[0] + acc[1]) * (1.f / 16777216.f);   // /(B*S*H) each
}

extern "C" void kernel_launch(void* const* d_in, const int* in_sizes, int n_in,
                              void* d_out, int out_size, void* d_ws, size_t ws_size,
                              hipStream_t stream) {
  (void)in_sizes; (void)n_in; (void)out_size; (void)ws_size;
  const float* X = (const float*)d_in[0];

  // ---- workspace carve ----
  bf16_t* wbf = (bf16_t*)d_ws;                                   // 16 * 2^20 bf16 = 32MB
  float* fws  = (float*)((char*)d_ws + ((size_t)16 << 20) * sizeof(bf16_t));
  const size_t SEQSZ = (size_t)BATCH * SEQ * HDIM;
  float* seqA = fws;  fws += SEQSZ;      // e1 / b1
  float* seqB = fws;  fws += SEQSZ;      // e2 / b2
  float* e3   = fws;  fws += SEQSZ;
  float* b3   = fws;  fws += SEQSZ;
  float* h0   = fws;  fws += (size_t)BATCH * HDIM;
  float* h1   = fws;  fws += (size_t)BATCH * HDIM;
  float* cbuf = fws;  fws += (size_t)BATCH * HDIM;
  float* dh0  = fws;  fws += (size_t)BATCH * HDIM;
  float* dh1  = fws;  fws += (size_t)BATCH * HDIM;
  float* pooled = fws; fws += (size_t)BATCH * HDIM;
  float* sqacc  = fws; fws += 2;

  // ---- convert all 8 layers' Wih/Whh to bf16 ----
  Ptrs16 wp;
  for (int L = 0; L < 8; ++L) {
    wp.p[2 * L + 0] = (const float*)d_in[1 + 4 * L + 0];
    wp.p[2 * L + 1] = (const float*)d_in[1 + 4 * L + 1];
  }
  hipLaunchKernelGGL(k_cvt_weights, dim3(2048), dim3(256), 0, stream, wp, wbf);
  hipLaunchKernelGGL(k_fill0, dim3(1), dim3(32), 0, stream, sqacc, 2);

  const bf16_t* Wih[8]; const bf16_t* Whh[8];
  const float*  Bih[8]; const float*  Bhh[8];
  for (int L = 0; L < 8; ++L) {
    Wih[L] = wbf + (size_t)(2 * L + 0) * ((size_t)1 << 20);
    Whh[L] = wbf + (size_t)(2 * L + 1) * ((size_t)1 << 20);
    Bih[L] = (const float*)d_in[1 + 4 * L + 2];
    Bhh[L] = (const float*)d_in[1 + 4 * L + 3];
  }

  dim3 sgrid(2, 8), sblk(256);
  auto step = [&](const StepArgs& sa) {
    hipLaunchKernelGGL(k_lstm_step, sgrid, sblk, 0, stream, sa);
  };

  // encoder layer: 128 sequential steps, h ping-pong, c in-place
  auto runEnc = [&](int L, const float* xb, bool xrev,
                    const float* rb, bool rrev, float* so) -> float* {
    hipLaunchKernelGGL(k_fill0, dim3(256), dim3(256), 0, stream, h0, BATCH * HDIM);
    hipLaunchKernelGGL(k_fill0, dim3(256), dim3(256), 0, stream, cbuf, BATCH * HDIM);
    float* hp = h0; float* hn = h1;
    for (int t = 0; t < SEQ; ++t) {
      StepArgs sa{};
      int xt = xrev ? (SEQ - 1 - t) : t;
      sa.x = xb + (size_t)xt * HDIM;  sa.xbs = (long long)SEQ * HDIM;
      sa.h_in = hp; sa.c_in = cbuf;
      sa.Wih = Wih[L]; sa.Whh = Whh[L]; sa.bih = Bih[L]; sa.bhh = Bhh[L];
      sa.h_out = hn; sa.c_out = cbuf;
      sa.seq_out = so + (size_t)t * HDIM; sa.seq_bs = (long long)SEQ * HDIM;
      if (rb) {
        int rt = rrev ? (SEQ - 1 - t) : t;
        sa.res = rb + (size_t)rt * HDIM; sa.res_bs = (long long)SEQ * HDIM;
      }
      step(sa);
      float* tmp = hp; hp = hn; hn = tmp;
    }
    return hp;   // final hidden state
  };

  // ---- forward stack ----
  runEnc(0, X,    false, X, false, seqA);                 // e1 = lstm(x)+x
  runEnc(1, seqA, false, X, false, seqB);                 // e2 = lstm(e1)+x
  float* hTf = runEnc(2, seqB, false, nullptr, false, e3);// e3 ; cbuf holds cT

  // forward decoder: x_{t+1} = h_{t+1} = h_new ; c in-place; sqerr vs x
  {
    float* prev = nullptr;
    for (int t = 0; t < SEQ; ++t) {
      StepArgs sa{};
      if (t == 0) {
        sa.x = e3 + (size_t)(SEQ - 1) * HDIM; sa.xbs = (long long)SEQ * HDIM;
        sa.h_in = hTf;
      } else {
        sa.x = prev; sa.xbs = HDIM;
        sa.h_in = prev;
      }
      sa.c_in = cbuf; sa.c_out = cbuf;
      sa.Wih = Wih[3]; sa.Whh = Whh[3]; sa.bih = Bih[3]; sa.bhh = Bhh[3];
      float* out = (t & 1) ? dh1 : dh0;
      sa.h_out = out;
      sa.ref = X + (size_t)t * HDIM; sa.ref_bs = (long long)SEQ * HDIM;
      sa.sq_accum = sqacc + 0;
      step(sa);
      prev = out;
    }
  }

  // ---- backward stack (xb = time-reversed x, handled via index) ----
  runEnc(4, X,    true,  X, true,  seqA);                 // b1 = lstm(xb)+xb
  runEnc(5, seqA, false, X, true,  seqB);                 // b2 = lstm(b1)+xb
  float* hTb = runEnc(6, seqB, false, nullptr, false, b3);// b3 ; cbuf holds cT

  // backward decoder: h_new = cell(x, h0, c0); x_{t+1}=h_new; c0 fixed
  {
    float* prev = nullptr;
    for (int t = 0; t < SEQ; ++t) {
      StepArgs sa{};
      if (t == 0) {
        sa.x = b3 + (size_t)(SEQ - 1) * HDIM; sa.xbs = (long long)SEQ * HDIM;
      } else {
        sa.x = prev; sa.xbs = HDIM;
      }
      sa.h_in = hTb; sa.c_in = cbuf; sa.c_out = nullptr;  // c0 never updated
      sa.Wih = Wih[7]; sa.Whh = Whh[7]; sa.bih = Bih[7]; sa.bhh = Bhh[7];
      float* out = (t & 1) ? dh1 : dh0;
      sa.h_out = out;
      sa.ref = X + (size_t)(SEQ - 1 - t) * HDIM; sa.ref_bs = (long long)SEQ * HDIM;
      sa.sq_accum = sqacc + 1;
      step(sa);
      prev = out;
    }
  }

  // ---- attention pooling (exact: softmax over singleton axis == 1) + output ----
  hipLaunchKernelGGL(k_pool, dim3(BATCH), dim3(256), 0, stream, X, e3, b3, pooled);
  hipLaunchKernelGGL(k_out,  dim3(BATCH), dim3(256), 0, stream,
                     pooled, (const float*)d_in[37], (const float*)d_in[38], (float*)d_out);
  hipLaunchKernelGGL(k_finalize, dim3(1), dim3(1), 0, stream, sqacc, (float*)d_out + 65536);
}
